// SpectralConvLaplace_14834817040474
// MI455X (gfx1250) — matrix operand order repaired and checked by reading, not compiled, not run on hardware
//
#include <hip/hip_runtime.h>
#include <hip/hip_bf16.h>
#include <math.h>

// ---------------- problem dims ----------------
#define BB   8
#define CIN  32
#define COUT 32
#define NX   256     // N1
#define NYS  256     // N2 (spatial)
#define F2V  129     // rfft bins along y
#define F2P  160     // padded bins (multiple of 32)
#define K1D  16
#define K2D  9
#define IKP  512     // CIN*K1D
#define TWO_PI 6.283185307179586f
#define INV_NORM (1.0f/65536.0f)
#define CDIV(a,b) (((a)+(b)-1)/(b))

typedef _Float16 v16h __attribute__((ext_vector_type(16)));
typedef _Float16 v8h  __attribute__((ext_vector_type(8)));
typedef float    v8f  __attribute__((ext_vector_type(8)));
typedef float    v4f  __attribute__((ext_vector_type(4)));
typedef unsigned int uv4 __attribute__((ext_vector_type(4)));
typedef int      iv4 __attribute__((ext_vector_type(4)));
typedef int      iv8 __attribute__((ext_vector_type(8)));

// =====================================================================
// Generic WMMA GEMM:  C[m,n] (f32) (+)= sign * sum_k A[m,k]*Bt[n,k]
//   A: f32, row addressing  A + ((m/aDiv)%aMod)*aImg + (m%aDiv)*lda
//   Bt: f16, [n][k] layout  Bt + ((m0/bDiv)%bMod)*bImg + n*ldb   (k contiguous)
//   C: f32                 C + ((m/cDiv)%cMod)*cImg + (m%cDiv)*cSM + n*cSN
// Requirements: M%16==0, N%(16*NT)==0, K%32==0, bDiv%16==0, lda%4==0, ldb%16==0.
// One wave computes a 16 x (16*NT) strip: the converted f32->f16 A fragment is
// reused across NT v_wmma_f32_16x16x32_f16 issues per K-step.
// =====================================================================
struct GemmP {
  const float* A;     long aDiv, aMod, aImg, lda;
  const _Float16* Bt; long bDiv, bMod, bImg, ldb;
  float* C;           long cDiv, cMod, cImg, cSM, cSN;
  int M, N, K; float sign; int accum;
};

template <int NT>
__global__ __launch_bounds__(256) void k_gemm(GemmP p) {
  const int lane = threadIdx.x & 31;
  const int wv   = threadIdx.x >> 5;
  const long nTg = p.N / (16 * NT);
  long tile = (long)blockIdx.x * 8 + wv;
  if (tile >= (long)(p.M >> 4) * nTg) return;
  const long mt = tile / nTg, ng = tile - mt * nTg;
  const int lm = lane & 15, lh = lane >> 4;

  const long aRow = mt * 16 + lm;
  const float* ap = p.A + ((aRow / p.aDiv) % p.aMod) * p.aImg
                        + (aRow % p.aDiv) * p.lda + (long)lh * 8;
  const long m0 = mt * 16;
  const _Float16* bp = p.Bt + ((m0 / p.bDiv) % p.bMod) * p.bImg
                            + (ng * 16 * NT + lm) * p.ldb + (long)lh * 8;

  v8f acc[NT];
#pragma unroll
  for (int j = 0; j < NT; ++j) acc[j] = (v8f){0.f,0.f,0.f,0.f,0.f,0.f,0.f,0.f};

  for (int k0 = 0; k0 < p.K; k0 += 32) {
    __builtin_prefetch(ap + k0 + 128, 0, 1);   // global_prefetch_b8
    v4f a0 = *(const v4f*)(ap + k0);
    v4f a1 = *(const v4f*)(ap + k0 + 4);
    v4f a2 = *(const v4f*)(ap + k0 + 16);
    v4f a3 = *(const v4f*)(ap + k0 + 20);
    v16h af;
    af[0]=(_Float16)a0[0];  af[1]=(_Float16)a0[1];  af[2]=(_Float16)a0[2];  af[3]=(_Float16)a0[3];
    af[4]=(_Float16)a1[0];  af[5]=(_Float16)a1[1];  af[6]=(_Float16)a1[2];  af[7]=(_Float16)a1[3];
    af[8]=(_Float16)a2[0];  af[9]=(_Float16)a2[1];  af[10]=(_Float16)a2[2]; af[11]=(_Float16)a2[3];
    af[12]=(_Float16)a3[0]; af[13]=(_Float16)a3[1]; af[14]=(_Float16)a3[2]; af[15]=(_Float16)a3[3];
#pragma unroll
    for (int j = 0; j < NT; ++j) {
      const _Float16* bj = bp + (long)j * 16 * p.ldb + k0;
      v8h b0 = *(const v8h*)(bj);
      v8h b1 = *(const v8h*)(bj + 16);
      v16h bf = __builtin_shufflevector(b0, b1, 0,1,2,3,4,5,6,7,8,9,10,11,12,13,14,15);
      acc[j] = __builtin_amdgcn_wmma_f32_16x16x32_f16(false, af, false, bf,
                                                      (short)0, acc[j], false, false);
    }
  }
#pragma unroll
  for (int j = 0; j < NT; ++j) {
    const long cn = ng * 16 * NT + j * 16 + lm;
#pragma unroll
    for (int r = 0; r < 8; ++r) {
      const long cm = mt * 16 + lh * 8 + r;
      const long offc = ((cm / p.cDiv) % p.cMod) * p.cImg
                      + (cm % p.cDiv) * p.cSM + cn * p.cSN;
      float v = acc[j][r] * p.sign;
      if (p.accum) v += p.C[offc];
      p.C[offc] = v;
    }
  }
}

static inline void gemm(hipStream_t s,
    const float* A, long aDiv,long aMod,long aImg,long lda,
    const _Float16* Bt, long bDiv,long bMod,long bImg,long ldb,
    float* C, long cDiv,long cMod,long cImg,long cSM,long cSN,
    int M,int N,int K,float sign,int accum) {
  GemmP p{A,aDiv,aMod,aImg,lda,Bt,bDiv,bMod,bImg,ldb,C,cDiv,cMod,cImg,cSM,cSN,M,N,K,sign,accum};
  if ((N & 63) == 0) {
    long tiles = (long)(M/16)*(N/64);
    k_gemm<4><<<(int)CDIV(tiles,8),256,0,s>>>(p);
  } else if ((N & 31) == 0) {
    long tiles = (long)(M/16)*(N/32);
    k_gemm<2><<<(int)CDIV(tiles,8),256,0,s>>>(p);
  } else {
    long tiles = (long)(M/16)*(N/16);
    k_gemm<1><<<(int)CDIV(tiles,8),256,0,s>>>(p);
  }
}

// ===================== table builders =====================
__global__ __launch_bounds__(256) void k_fill(float* p, long n, float v) {
  long i = (long)blockIdx.x*256 + threadIdx.x; if (i < n) p[i] = v;
}
// forward y-DFT tables Bt[f2][y], f2<129 valid, zero padded to 160
__global__ __launch_bounds__(256) void k_dfty(_Float16* C, _Float16* S) {
  int i = blockIdx.x*256 + threadIdx.x; if (i >= F2P*NYS) return;
  int y = i & 255, n = i >> 8;
  float c = 0.f, sn = 0.f;
  if (n < F2V) { int ph = (n*y) & 255; float th = ph*(TWO_PI/256.0f);
                 c = __cosf(th); sn = -__sinf(th); }
  C[i] = (_Float16)c; S[i] = (_Float16)sn;
}
// x-DFT tables (forward: cos,-sin ; inverse: cos,+sin), [a][b] 256x256
__global__ __launch_bounds__(256) void k_dftx(_Float16* WC,_Float16* WS,_Float16* IC,_Float16* IS) {
  int i = blockIdx.x*256 + threadIdx.x; if (i >= 65536) return;
  int b_ = i & 255, a_ = i >> 8;
  int ph = (a_*b_) & 255; float th = ph*(TWO_PI/256.0f);
  float c = __cosf(th), sn = __sinf(th);
  WC[i]=(_Float16)c; WS[i]=(_Float16)(-sn); IC[i]=(_Float16)c; IS[i]=(_Float16)sn;
}
// inverse y (real output, Hermitian weights) Bt[y][f2], zero-padded f2>=129
__global__ __launch_bounds__(256) void k_idfty(_Float16* C, _Float16* S) {
  int i = blockIdx.x*256 + threadIdx.x; if (i >= NYS*F2P) return;
  int f2 = i % F2P, y = i / F2P;
  float c = 0.f, sn = 0.f;
  if (f2 < F2V) { float w = (f2==0 || f2==128) ? 1.f : 2.f;
                  int ph = (f2*y) & 255; float th = ph*(TWO_PI/256.0f);
                  c = w*__cosf(th); sn = -w*__sinf(th); }
  C[i]=(_Float16)c; S[i]=(_Float16)sn;
}
// P1 = 1/(i*w1 - p1): write f16 [io][k][f1] and f16 [io][f1][k32] (k>=16 zero)
__global__ __launch_bounds__(256) void k_P1(const float* pr, const float* pi,
    _Float16* kR, _Float16* kI, _Float16* tR, _Float16* tI) {
  long i = (long)blockIdx.x*256 + threadIdx.x;
  if (i >= (long)CIN*COUT*NX*32) return;
  int k = i & 31; int f1 = (i >> 5) & 255; long io = i >> 13;
  float PR = 0.f, PI = 0.f;
  if (k < K1D) {
    float f = (f1 < 128) ? (float)f1 : (float)(f1 - 256);
    float w1 = TWO_PI * f;
    float dr = -pr[io*K1D + k], di = w1 - pi[io*K1D + k];
    float den = 1.0f / (dr*dr + di*di);
    PR = dr * den; PI = -di * den;
    kR[(io*K1D + k)*NX + f1] = (_Float16)PR;
    kI[(io*K1D + k)*NX + f1] = (_Float16)PI;
  }
  tR[i] = (_Float16)PR; tI[i] = (_Float16)PI;
}
// P2 f32 [io][l16][160], zeros l>=9 / f2>=129
__global__ __launch_bounds__(256) void k_P2(const float* pr, const float* pi,
                                            float* R, float* I) {
  long i = (long)blockIdx.x*256 + threadIdx.x;
  if (i >= (long)CIN*COUT*16*F2P) return;
  int f2 = i % F2P; int l = (i / F2P) & 15; long io = i / (16*F2P);
  float PR = 0.f, PI = 0.f;
  if (l < K2D && f2 < F2V) {
    float w2 = TWO_PI * (float)f2;
    float dr = -pr[io*K2D + l], di = w2 - pi[io*K2D + l];
    float den = 1.0f / (dr*dr + di*di);
    PR = dr * den; PI = -di * den;
  }
  R[i] = PR; I[i] = PI;
}
// G1[io][f2][k32] = sum_l residue[io,k,l] * P2[io,l,f2]   (f32, zero k>=16)
__global__ __launch_bounds__(256) void k_G1(const float* rR, const float* rI,
    const float* P2R, const float* P2I, float* GR, float* GI) {
  long i = (long)blockIdx.x*256 + threadIdx.x;
  if (i >= (long)CIN*COUT*F2P*32) return;
  int k = i & 31; int f2 = (i >> 5) % F2P; long io = i / (32*F2P);
  float gr = 0.f, gi = 0.f;
  if (k < K1D) {
    for (int l = 0; l < K2D; ++l) {
      float ar = rR[(io*K1D + k)*K2D + l], ai = rI[(io*K1D + k)*K2D + l];
      float p2r_ = P2R[(io*16 + l)*F2P + f2], p2i_ = P2I[(io*16 + l)*F2P + f2];
      gr += ar*p2r_ - ai*p2i_;
      gi += ar*p2i_ + ai*p2r_;
    }
  }
  GR[i] = gr; GI[i] = gi;
}
// E1T[o][z][i*16+k] = exp(p1 * z/256)
__global__ __launch_bounds__(256) void k_E1(const float* pr, const float* pi,
                                            float* R, float* I) {
  long idx = (long)blockIdx.x*256 + threadIdx.x;
  if (idx >= (long)COUT*NX*IKP) return;
  int ik = idx & 511; int z = (idx >> 9) & 255; int o = idx >> 17;
  int i = ik >> 4, k = ik & 15;
  float t = (float)z / 256.0f;
  long p = ((long)i*COUT + o)*K1D + k;
  float e = __expf(pr[p]*t), th = pi[p]*t;
  R[idx] = e*__cosf(th); I[idx] = e*__sinf(th);
}
// E2[io][l][w] = exp(p2 * w/256)
__global__ __launch_bounds__(256) void k_E2(const float* pr, const float* pi,
                                            float* R, float* I) {
  long idx = (long)blockIdx.x*256 + threadIdx.x;
  if (idx >= (long)CIN*COUT*K2D*NYS) return;
  int w = idx & 255; int l = (idx >> 8) % K2D; long io = idx / (K2D*256);
  float t = (float)w / 256.0f;
  float e = __expf(pr[io*K2D + l]*t), th = pi[io*K2D + l]*t;
  R[idx] = e*__cosf(th); I[idx] = e*__sinf(th);
}
__global__ __launch_bounds__(256) void k_bias(float* out, const float* bias) {
  long i = (long)blockIdx.x*256 + threadIdx.x;
  if (i < (long)BB*COUT*NX*NYS) out[i] = bias[(i >> 16) & 31];
}

// ============ transpose+cvt alpha[b,i,f2,f1] -> alphaT16[b,i,f1,f2] ============
// 32x32 f32 tiles staged to LDS via the Tensor Data Mover (TDM).
#if defined(__has_builtin)
#if __has_builtin(__builtin_amdgcn_tensor_load_to_lds) && __has_builtin(__builtin_amdgcn_s_wait_tensorcnt)
#define USE_TDM 1
#endif
#endif
__global__ __launch_bounds__(256) void k_transpose_cvt(const float* __restrict__ src,
                                                       _Float16* __restrict__ dst) {
  __shared__ float tile[1024];           // 32 (f2) x 32 (f1)
  const int img = blockIdx.x / 40;
  const int t   = blockIdx.x % 40;
  const int tf2 = t / 8, tf1 = t % 8;
  const float* g = src + ((long)img*F2P + tf2*32)*NX + tf1*32;
#ifdef USE_TDM
  if (threadIdx.x == 0) {
    unsigned long long ga = (unsigned long long)(size_t)g;
    unsigned int lds = (unsigned int)(size_t)(&tile[0]);
    uv4 g0; g0[0] = 1u;                       // count=1, user descriptor
    g0[1] = lds;                              // lds_addr [63:32]
    g0[2] = (unsigned int)(ga & 0xFFFFFFFFull);            // global_addr lo
    g0[3] = (unsigned int)((ga >> 32) & 0x01FFFFFFull) | (2u << 30); // addr hi | type=2
    iv8 g1;
    g1[0] = (2 << 16);                        // data_size = 4B
    g1[1] = (NX & 0xFFFF) << 16;              // tensor_dim0 = 256
    g1[2] = (F2P & 0xFFFF) << 16;             // tensor_dim1 = 160
    g1[3] = (32 << 16);                       // tile_dim0  = 32
    g1[4] = 32;                               // tile_dim1  = 32
    g1[5] = NX;                               // tensor_dim0_stride = 256
    g1[6] = 0; g1[7] = 0;
    iv4 z4 = {0,0,0,0};
#if __clang_major__ >= 23
    iv8 z8 = {0,0,0,0,0,0,0,0};
    __builtin_amdgcn_tensor_load_to_lds(g0, g1, z4, z4, z8, 0);
#else
    __builtin_amdgcn_tensor_load_to_lds(g0, g1, z4, z4, 0);
#endif
    __builtin_amdgcn_s_wait_tensorcnt(0);
  }
  __syncthreads();
#else
  for (int e = threadIdx.x; e < 1024; e += 256)
    tile[e] = g[(e >> 5) * NX + (e & 31)];
  __syncthreads();
#endif
  const int l1 = threadIdx.x & 31;             // f1 local
  const int l2b = (threadIdx.x >> 5) * 4;      // f2 local base
#pragma unroll
  for (int j = 0; j < 4; ++j) {
    int l2 = l2b + j;
    dst[((long)img*NX + tf1*32 + l1)*F2P + tf2*32 + l2] = (_Float16)tile[l2*32 + l1];
  }
}

// ============ r1 accumulation (Hadamard over (f2,f1), reduce over i) ============
__global__ __launch_bounds__(256) void k_accum_r1(const float* aR, const float* aI,
    const float* hR, const float* hI, float* r1R, float* r1I, int o) {
  long idx = (long)blockIdx.x*256 + threadIdx.x;
  if (idx >= (long)BB*F2P*NX) return;
  int f1 = idx & 255; int f2 = (idx >> 8) % F2P; int b = (int)(idx / (F2P*256));
  float sr = 0.f, si = 0.f;
  for (int i = 0; i < CIN; ++i) {
    long ao = (((long)(b*CIN + i))*F2P + f2)*NX + f1;
    long ho = ((long)i*F2P + f2)*NX + f1;
    float ar = aR[ao], ai = aI[ao], br = hR[ho], bi = hI[ho];
    sr += ar*br - ai*bi;
    si += ar*bi + ai*br;
  }
  long ro = (((long)(b*COUT + o))*F2P + f2)*NX + f1;
  r1R[ro] += sr; r1I[ro] += si;
}

// r2 = -residue * S
__global__ __launch_bounds__(256) void k_finish_r2(const float* SR, const float* SI,
    const float* rR, const float* rI, float* oR, float* oI) {
  long idx = (long)blockIdx.x*256 + threadIdx.x;
  if (idx >= (long)BB*CIN*COUT*K1D*K2D) return;
  int l = idx % K2D; int k = (int)((idx / K2D) % K1D); long bio = idx / (K1D*K2D);
  long io = bio % ((long)CIN*COUT);
  long so = (bio*16 + l)*16 + k;
  float sr = SR[so], si = SI[so];
  float ar = rR[(io*K1D + k)*K2D + l], ai = rI[(io*K1D + k)*K2D + l];
  long ro = (bio*16 + k)*16 + l;
  oR[ro] = -(ar*sr - ai*si);
  oI[ro] = -(ar*si + ai*sr);
}

// U16[(b,o)][w][i*16+k] = sum_l r2[b,i,o,k,l] * E2[i,o,l,w]  (f16 Bt layout)
__global__ __launch_bounds__(256) void k_build_U(const float* r2R, const float* r2I,
    const float* E2R, const float* E2I, _Float16* UR, _Float16* UI) {
  long idx = (long)blockIdx.x*256 + threadIdx.x;
  if (idx >= (long)BB*COUT*NYS*IKP) return;
  int ik = idx & 511; int w = (idx >> 9) & 255; int bo = idx >> 17;
  int b = bo >> 5, o = bo & 31, i = ik >> 4, k = ik & 15;
  float ur = 0.f, ui = 0.f;
  long rbase = (((long)(b*CIN + i)*COUT + o)*16 + k)*16;
  long ebase = ((long)i*COUT + o)*K2D;
  for (int l = 0; l < K2D; ++l) {
    float rr = r2R[rbase + l], ri = r2I[rbase + l];
    float er = E2R[(ebase + l)*256 + w], ei = E2I[(ebase + l)*256 + w];
    ur += rr*er - ri*ei;
    ui += rr*ei + ri*er;
  }
  UR[idx] = (_Float16)ur; UI[idx] = (_Float16)ui;
}

// =====================================================================
extern "C" void kernel_launch(void* const* d_in, const int* in_sizes, int n_in,
                              void* d_out, int out_size, void* d_ws, size_t ws_size,
                              hipStream_t stream) {
  const float* x    = (const float*)d_in[0];
  const float* resR = (const float*)d_in[1];
  const float* resI = (const float*)d_in[2];
  const float* p1r  = (const float*)d_in[3];
  const float* p1i  = (const float*)d_in[4];
  const float* p2r  = (const float*)d_in[5];
  const float* p2i  = (const float*)d_in[6];
  const float* bias = (const float*)d_in[7];
  float* out = (float*)d_out;

  char* W = (char*)d_ws;
  size_t off = 0;
  auto take = [&](size_t bytes) -> char* {
    char* r = W + off; off = (off + bytes + 255) & ~(size_t)255; return r;
  };
  const size_t AEL = (size_t)BB*CIN*F2P*NX;   // 10,485,760
  float*    alphaR = (float*)take(AEL*4);
  float*    alphaI = (float*)take(AEL*4);
  _Float16* aT16R  = (_Float16*)take(AEL*2);
  _Float16* aT16I  = (_Float16*)take(AEL*2);
  char*     big    = take((size_t)268435456);           // tmpA -> s1 -> T -> U16
  float*    tmpAR  = (float*)big;
  float*    tmpAI  = (float*)(big + 41943040);
  float*    s1R = tmpAR, *s1I = tmpAI;
  float*    TR  = (float*)big;
  float*    TI  = (float*)(big + 134217728);
  _Float16* UR  = (_Float16*)big;
  _Float16* UI  = (_Float16*)(big + 67108864);
  float*    r1R = (float*)take(AEL*4);
  float*    r1I = (float*)take(AEL*4);
  float*    SR  = (float*)take((size_t)2097152*4);
  float*    SI  = (float*)take((size_t)2097152*4);
  float*    r2R = (float*)take((size_t)2097152*4);
  float*    r2I = (float*)take((size_t)2097152*4);
  float*    HwR = (float*)take((size_t)1310720*4);
  float*    HwI = (float*)take((size_t)1310720*4);
  _Float16* DyC = (_Float16*)take(40960*2);
  _Float16* DyS = (_Float16*)take(40960*2);
  _Float16* WxC = (_Float16*)take(65536*2);
  _Float16* WxS = (_Float16*)take(65536*2);
  _Float16* IxC = (_Float16*)take(65536*2);
  _Float16* IxS = (_Float16*)take(65536*2);
  _Float16* IyC = (_Float16*)take(40960*2);
  _Float16* IyS = (_Float16*)take(40960*2);
  float*    P2R = (float*)take((size_t)2621440*4);
  float*    P2I = (float*)take((size_t)2621440*4);
  _Float16* P1kR = (_Float16*)take((size_t)2097152*2);  // [io][k][f1]
  _Float16* P1kI = (_Float16*)take((size_t)2097152*2);
  _Float16* P1tR = (_Float16*)take((size_t)8388608*2);  // [io][f1][k32]
  _Float16* P1tI = (_Float16*)take((size_t)8388608*2);
  float*    G1R = (float*)take((size_t)5242880*4);      // [io][f2][k32]
  float*    G1I = (float*)take((size_t)5242880*4);
  float*    E1R = (float*)take((size_t)4194304*4);      // [o][z][ik]
  float*    E1I = (float*)take((size_t)4194304*4);
  float*    E2R = (float*)take((size_t)2359296*4);      // [io][l][w]
  float*    E2I = (float*)take((size_t)2359296*4);
  (void)in_sizes; (void)n_in; (void)out_size; (void)ws_size;

  hipStream_t s = stream;
  // ---------------- tables ----------------
  k_dfty <<<CDIV(40960,256),256,0,s>>>(DyC, DyS);
  k_dftx <<<CDIV(65536,256),256,0,s>>>(WxC, WxS, IxC, IxS);
  k_idfty<<<CDIV(40960,256),256,0,s>>>(IyC, IyS);
  k_P1<<<CDIV((long)CIN*COUT*NX*32,256),256,0,s>>>(p1r,p1i,P1kR,P1kI,P1tR,P1tI);
  k_P2<<<CDIV((long)CIN*COUT*16*F2P,256),256,0,s>>>(p2r,p2i,P2R,P2I);
  k_G1<<<CDIV((long)CIN*COUT*F2P*32,256),256,0,s>>>(resR,resI,P2R,P2I,G1R,G1I);
  k_E1<<<CDIV((long)COUT*NX*IKP,256),256,0,s>>>(p1r,p1i,E1R,E1I);
  k_E2<<<CDIV((long)CIN*COUT*K2D*NYS,256),256,0,s>>>(p2r,p2i,E2R,E2I);
  k_bias<<<CDIV((long)BB*COUT*NX*NYS,256),256,0,s>>>(out, bias);

  // ---- C1: y-DFT, tmpA[b,i,f2,x] (transposed store per image) ----
  {
    int M = BB*CIN*NX;
    gemm(s, x,     M,1,0,NYS,  DyC, M,1,0,NYS,
         tmpAR, 256, (long)BB*CIN, (long)F2P*NX, 1, NX, M, F2P, NYS, 1.f, 0);
    gemm(s, x,     M,1,0,NYS,  DyS, M,1,0,NYS,
         tmpAI, 256, (long)BB*CIN, (long)F2P*NX, 1, NX, M, F2P, NYS, 1.f, 0);
  }
  // ---- C2: x-DFT -> alpha[b,i,f2,f1] ----
  {
    int M = BB*CIN*F2P;
    gemm(s, tmpAR, M,1,0,NX, WxC, M,1,0,NX, alphaR, M,1,0,NX,1, M,NX,NX,  1.f,0);
    gemm(s, tmpAI, M,1,0,NX, WxS, M,1,0,NX, alphaR, M,1,0,NX,1, M,NX,NX, -1.f,1);
    gemm(s, tmpAR, M,1,0,NX, WxS, M,1,0,NX, alphaI, M,1,0,NX,1, M,NX,NX,  1.f,0);
    gemm(s, tmpAI, M,1,0,NX, WxC, M,1,0,NX, alphaI, M,1,0,NX,1, M,NX,NX,  1.f,1);
  }
  // ---- transpose + f16 cvt: alphaT16[b,i,f1,f2] (TDM-staged tiles) ----
  k_transpose_cvt<<<BB*CIN*40,256,0,s>>>(alphaR, aT16R);
  k_transpose_cvt<<<BB*CIN*40,256,0,s>>>(alphaI, aT16I);

  // ---- r1 path: per-o Hw then i-reduction ----
  k_fill<<<CDIV((long)AEL,256),256,0,s>>>(r1R, (long)AEL, 0.f);
  k_fill<<<CDIV((long)AEL,256),256,0,s>>>(r1I, (long)AEL, 0.f);
  for (int o = 0; o < COUT; ++o) {
    const float*    gR = G1R + (long)o*F2P*32;
    const float*    gI = G1I + (long)o*F2P*32;
    const _Float16* pR = P1tR + (long)o*NX*32;
    const _Float16* pI = P1tI + (long)o*NX*32;
    int M = CIN*F2P;
    long aImg = (long)COUT*F2P*32, bImg = (long)COUT*NX*32;
    gemm(s, gR, F2P,CIN,aImg,32, pR, F2P,CIN,bImg,32, HwR, M,1,0,NX,1, M,NX,32,  1.f,0);
    gemm(s, gI, F2P,CIN,aImg,32, pI, F2P,CIN,bImg,32, HwR, M,1,0,NX,1, M,NX,32, -1.f,1);
    gemm(s, gR, F2P,CIN,aImg,32, pI, F2P,CIN,bImg,32, HwI, M,1,0,NX,1, M,NX,32,  1.f,0);
    gemm(s, gI, F2P,CIN,aImg,32, pR, F2P,CIN,bImg,32, HwI, M,1,0,NX,1, M,NX,32,  1.f,1);
    k_accum_r1<<<CDIV((long)BB*F2P*NX,256),256,0,s>>>(alphaR,alphaI,HwR,HwI,r1R,r1I,o);
  }
  // ---- C8: inverse x-DFT -> s1[b,o,x,f2] (transposed store) ----
  {
    int M = BB*COUT*F2P;
    long cImg = (long)NX*F2P;
    gemm(s, r1R, M,1,0,NX, IxC, M,1,0,NX, s1R, F2P,(long)BB*COUT,cImg,1,F2P, M,NX,NX,  1.f,0);
    gemm(s, r1I, M,1,0,NX, IxS, M,1,0,NX, s1R, F2P,(long)BB*COUT,cImg,1,F2P, M,NX,NX, -1.f,1);
    gemm(s, r1R, M,1,0,NX, IxS, M,1,0,NX, s1I, F2P,(long)BB*COUT,cImg,1,F2P, M,NX,NX,  1.f,0);
    gemm(s, r1I, M,1,0,NX, IxC, M,1,0,NX, s1I, F2P,(long)BB*COUT,cImg,1,F2P, M,NX,NX,  1.f,1);
  }
  // ---- C9: inverse y (real), accumulate x1/65536 into out ----
  {
    int M = BB*COUT*NX;
    gemm(s, s1R, M,1,0,F2P, IyC, M,1,0,F2P, out, M,1,0,NYS,1, M,NYS,F2P, INV_NORM,1);
    gemm(s, s1I, M,1,0,F2P, IyS, M,1,0,F2P, out, M,1,0,NYS,1, M,NYS,F2P, INV_NORM,1);
  }
  // ---- C3: T[b,i,o,l,f1] = sum_f2 P2 * alpha ----
  {
    int M = BB*CIN*COUT*K1D;
    long aImg = (long)16*F2P, bImg = (long)NX*F2P;
    gemm(s, P2R, 16,(long)CIN*COUT,aImg,F2P, aT16R, 512,(long)BB*CIN,bImg,F2P, TR, M,1,0,NX,1, M,NX,F2P,  1.f,0);
    gemm(s, P2I, 16,(long)CIN*COUT,aImg,F2P, aT16I, 512,(long)BB*CIN,bImg,F2P, TR, M,1,0,NX,1, M,NX,F2P, -1.f,1);
    gemm(s, P2R, 16,(long)CIN*COUT,aImg,F2P, aT16I, 512,(long)BB*CIN,bImg,F2P, TI, M,1,0,NX,1, M,NX,F2P,  1.f,0);
    gemm(s, P2I, 16,(long)CIN*COUT,aImg,F2P, aT16R, 512,(long)BB*CIN,bImg,F2P, TI, M,1,0,NX,1, M,NX,F2P,  1.f,1);
  }
  // ---- C4: S[b,i,o,l,k] = sum_f1 T * P1 ----
  {
    int M = BB*CIN*COUT*K1D;
    long bImg = (long)K1D*NX;
    gemm(s, TR, M,1,0,NX, P1kR, 16,(long)CIN*COUT,bImg,NX, SR, M,1,0,16,1, M,16,NX,  1.f,0);
    gemm(s, TI, M,1,0,NX, P1kI, 16,(long)CIN*COUT,bImg,NX, SR, M,1,0,16,1, M,16,NX, -1.f,1);
    gemm(s, TR, M,1,0,NX, P1kI, 16,(long)CIN*COUT,bImg,NX, SI, M,1,0,16,1, M,16,NX,  1.f,0);
    gemm(s, TI, M,1,0,NX, P1kR, 16,(long)CIN*COUT,bImg,NX, SI, M,1,0,16,1, M,16,NX,  1.f,1);
  }
  k_finish_r2<<<CDIV((long)BB*CIN*COUT*K1D*K2D,256),256,0,s>>>(SR,SI,resR,resI,r2R,r2I);
  // ---- C5: U (f16 Bt layout) — overwrites T region (T is dead) ----
  k_build_U<<<CDIV((long)BB*COUT*NYS*IKP,256),256,0,s>>>(r2R,r2I,E2R,E2I,UR,UI);
  // ---- C6: x2 = Re(E1 @ U), accumulate /65536 into out ----
  {
    int M = BB*COUT*NX;
    long aImg = (long)NX*IKP, bImg = (long)NX*IKP;
    gemm(s, E1R, 256,COUT,aImg,IKP, UR, 256,(long)BB*COUT,bImg,IKP, out, M,1,0,NYS,1, M,NYS,IKP,  INV_NORM,1);
    gemm(s, E1I, 256,COUT,aImg,IKP, UI, 256,(long)BB*COUT,bImg,IKP, out, M,1,0,NYS,1, M,NYS,IKP, -INV_NORM,1);
  }
}
